// RelativeMultiHeadAttention_48713519071494
// MI455X (gfx1250) — matrix operand order repaired and checked by reading, not compiled, not run on hardware
//
#include <hip/hip_runtime.h>
#include <hip/hip_bf16.h>

typedef __attribute__((ext_vector_type(16))) _Float16 v16h;
typedef __attribute__((ext_vector_type(8)))  _Float16 v8h;
typedef __attribute__((ext_vector_type(8)))  float    v8f;

namespace {
constexpr int kB    = 256;   // batch
constexpr int kT    = 128;   // sequence length
constexpr int kC    = 144;   // n_embd = H*hs
constexpr int kH    = 6;     // heads
constexpr int kHS   = 24;    // head size
constexpr int kKpad = 160;   // 144 padded up to multiple of 32
constexpr int kLP   = 136;   // padded row stride (16B-aligned for b128 LDS loads)
constexpr float kScale = 0.20412414523193154f; // 1/sqrt(24)
}

// ---- WMMA fragment loaders (CDNA5 16x16x32 f16 layouts, wave32) ----
// A fragment: element (m,k) at p[m*ld + k]; per lane the 16 halves are two
// contiguous 8-half runs at k = half*8 and k = 16 + half*8  ->  2 x b128.
__device__ inline v16h ldsA(const _Float16* p, int ld) {
  const int lane = threadIdx.x & 31;
  const _Float16* r = p + (lane & 15) * ld + ((lane >> 4) << 3);
  v8h lo = *(const v8h*)(r);
  v8h hi = *(const v8h*)(r + 16);
  return __builtin_shufflevector(lo, hi, 0, 1, 2, 3, 4, 5, 6, 7,
                                         8, 9, 10, 11, 12, 13, 14, 15);
}

// B fragment from N-major (B^T row-major) storage: element (k,n) at p[n*ld + k];
// per lane: two contiguous 8-half runs at k = half*8 and k = 16 + half*8  ->  2 x b128.
__device__ inline v16h ldsB(const _Float16* p, int ld) {
  const int lane = threadIdx.x & 31;
  const _Float16* r = p + (lane & 15) * ld + ((lane >> 4) << 3);
  v8h lo = *(const v8h*)(r);
  v8h hi = *(const v8h*)(r + 16);
  return __builtin_shufflevector(lo, hi, 0, 1, 2, 3, 4, 5, 6, 7,
                                         8, 9, 10, 11, 12, 13, 14, 15);
}

// D (16x16 f32): lane gives n = lane&15, VGPR r gives m = r + (lane>>4)*8.
__device__ inline void store_tile_f16(_Float16* p, int ld, const v8f& c) {
  const int lane = threadIdx.x & 31;
  const int col  = lane & 15;
  const int half = lane >> 4;
#pragma unroll
  for (int r = 0; r < 8; ++r) p[(r + half * 8) * ld + col] = (_Float16)c[r];
}

__device__ inline void store_tile_f32_scaled(float* p, int ld, const v8f& c, float mul) {
  const int lane = threadIdx.x & 31;
  const int col  = lane & 15;
  const int half = lane >> 4;
#pragma unroll
  for (int r = 0; r < 8; ++r) p[(r + half * 8) * ld + col] = c[r] * mul;
}

#define WMMA_F16(a, b, c) \
  __builtin_amdgcn_wmma_f32_16x16x32_f16(false, (a), false, (b), (short)0, (c), false, false)

// ---------------- Kernel 1: QKV projections (x @ Wq/Wk/Wv -> f16) ----------------
// One block per batch b. LDS: x (128x160 f16) + 3 transposed weights (144x160 f16).
__global__ void qkv_kernel(const float* __restrict__ x,
                           const float* __restrict__ Wq,
                           const float* __restrict__ Wk,
                           const float* __restrict__ Wv,
                           _Float16* __restrict__ qkv) {  // [3][B][T][C] f16
  extern __shared__ __align__(16) char smem[];
  _Float16* Xl = (_Float16*)smem;            // 128 x 160 (row-major, k-padded)
  _Float16* Wt = Xl + kT * kKpad;            // 3 x (144 n-rows x 160 k), N-major

  const int b = blockIdx.x;
  const int tid = threadIdx.x;
  const float* xb = x + (size_t)b * kT * kC;

  for (int i = tid; i < kT * kKpad; i += blockDim.x) {
    int m = i / kKpad, c = i % kKpad;
    Xl[i] = (c < kC) ? (_Float16)xb[m * kC + c] : (_Float16)0.0f;
  }
#pragma unroll
  for (int w = 0; w < 3; ++w) {
    const float* wg = (w == 0) ? Wq : (w == 1) ? Wk : Wv;
    _Float16* wt = Wt + w * (kC * kKpad);
    for (int i = tid; i < kC * kKpad; i += blockDim.x) {
      int n = i / kKpad, k = i % kKpad;
      wt[i] = (k < kC) ? (_Float16)wg[k * kC + n] : (_Float16)0.0f;  // transpose
    }
  }
  __syncthreads();

  const int wave = tid >> 5;
  for (int it = 0; it < 27; ++it) {           // 216 tiles = 8 waves * 27
    int tile = wave * 27 + it;
    int which = tile / 72;
    int rem = tile % 72;
    int mt = rem / 9, nt = rem % 9;
    const _Float16* wt = Wt + which * (kC * kKpad) + nt * 16 * kKpad;
    v8f c = {};
#pragma unroll
    for (int ks = 0; ks < 5; ++ks) {
      v16h a  = ldsA(Xl + mt * 16 * kKpad + ks * 32, kKpad);
      v16h bb = ldsB(wt + ks * 32, kKpad);
      c = WMMA_F16(a, bb, c);
    }
    _Float16* o = qkv + (size_t)which * ((size_t)kB * kT * kC)
                      + (size_t)b * kT * kC + mt * 16 * kC + nt * 16;
    store_tile_f16(o, kC, c);
  }
}

// ---------------- Kernel 2: attention per (b,h) ----------------
// LDS: Ql/Kl/Rl (128x32 f16), Vt (32x128 f16, transposed), S & P (128x136 f32),
//      A (128x136 f16).
__global__ void attn_kernel(const _Float16* __restrict__ Q,
                            const _Float16* __restrict__ K,
                            const _Float16* __restrict__ V,
                            const float* __restrict__ rel_table,
                            _Float16* __restrict__ Ctx) {
  extern __shared__ __align__(16) char smem[];
  _Float16* Ql = (_Float16*)smem;           // 128 x 32
  _Float16* Kl = Ql + kT * 32;              // 128 x 32   (N-major view of K^T)
  _Float16* Rl = Kl + kT * 32;              // 128 x 32 : Rl[j][d] = rel_table[127+j][d]
  _Float16* Vt = Rl + kT * 32;              // 32 x 128 : Vt[d][s] (N-major for ctx GEMM)
  float*    Pl = (float*)(Vt + 32 * kT);    // 128 x 136 : P[t][j] = q_t . R[j]
  float*    Sl = Pl + kT * kLP;             // 128 x 136 : S[t][s] = scale * q_t . k_s
  _Float16* Al = (_Float16*)(Sl + kT * kLP);// 128 x 136 : softmaxed attn (f16)

  const int blk = blockIdx.x;
  const int b = blk / kH, h = blk % kH;
  const int tid = threadIdx.x;

  // Raw-view layout: row n = h*128+t of the (B, 768, 24) reinterpretation.
  const size_t base = (size_t)b * (kT * kC) + (size_t)h * kT * kHS;
  for (int i = tid; i < kT * 32; i += blockDim.x) {
    int t = i >> 5, d = i & 31;
    _Float16 z = (_Float16)0.0f;
    bool ok = d < kHS;
    Ql[i] = ok ? Q[base + t * kHS + d] : z;
    Kl[i] = ok ? K[base + t * kHS + d] : z;
    Rl[i] = ok ? (_Float16)rel_table[(127 + t) * kHS + d] : z;
  }
  for (int i = tid; i < 32 * kT; i += blockDim.x) {
    int d = i >> 7, s = i & 127;                    // Vt[d][s]
    Vt[i] = (d < kHS) ? V[base + s * kHS + d] : (_Float16)0.0f;
  }
  __syncthreads();

  const int wave = tid >> 5;
  // 64 P tiles + 64 S tiles = 128 tiles over 8 waves. Single K-step (hs=24 <= 32).
  for (int it = 0; it < 16; ++it) {
    int tile = wave * 16 + it;
    bool isS = tile >= 64;
    int t64 = tile & 63;
    int tt = t64 >> 3, ct = t64 & 7;
    const _Float16* bp = Kl + (isS ? 0 : kT * 32) + ct * 16 * 32;  // Kl or Rl
    v8f c = {};
    v16h a  = ldsA(Ql + tt * 16 * 32, 32);
    v16h bb = ldsB(bp, 32);
    c = WMMA_F16(a, bb, c);
    if (isS) store_tile_f32_scaled(Sl + tt * 16 * kLP + ct * 16, kLP, c, kScale);
    else     store_tile_f32_scaled(Pl + tt * 16 * kLP + ct * 16, kLP, c, 1.0f);
  }
  __syncthreads();

  // Causal softmax with Toeplitz gather: logits[t][s] = S[t][s] + P[t][t-s], s<=t.
  if (tid < kT) {
    const int t = tid;
    float mx = -3.0e38f;
    for (int s = 0; s <= t; ++s) {
      float v = Sl[t * kLP + s] + Pl[t * kLP + (t - s)];
      mx = fmaxf(mx, v);
    }
    float sum = 0.0f;
    for (int s = 0; s <= t; ++s) {
      float e = __expf(Sl[t * kLP + s] + Pl[t * kLP + (t - s)] - mx);
      sum += e;
      Al[t * kLP + s] = (_Float16)e;
    }
    float inv = 1.0f / sum;
    for (int s = 0; s <= t; ++s)
      Al[t * kLP + s] = (_Float16)((float)Al[t * kLP + s] * inv);
    for (int s = t + 1; s < kT; ++s) Al[t * kLP + s] = (_Float16)0.0f;
  }
  __syncthreads();

  // ctx = A(128x128) @ V(128x24, zero-padded to 32). 8 mt x 2 nt tiles, 4 K-steps.
  for (int it = 0; it < 2; ++it) {
    int tile = wave * 2 + it;
    int mt = tile >> 1, nt = tile & 1;
    v8f c = {};
#pragma unroll
    for (int ks = 0; ks < 4; ++ks) {
      v16h a  = ldsA(Al + mt * 16 * kLP + ks * 32, kLP);
      v16h bb = ldsB(Vt + nt * 16 * kT + ks * 32, kT);
      c = WMMA_F16(a, bb, c);
    }
    const int lane = tid & 31;
    const int col  = (lane & 15) + nt * 16;
    const int half = lane >> 4;
    if (col < kHS) {  // drop the zero-padded d columns
#pragma unroll
      for (int r = 0; r < 8; ++r) {
        int t = mt * 16 + r + half * 8;
        // transpose(0,2,1,3): write at (b, t, h*hs + d)
        Ctx[(size_t)b * kT * kC + (size_t)t * kC + h * kHS + col] = (_Float16)c[r];
      }
    }
  }
}

// ---------------- Kernel 3: output projection (+bias, fp32 out) ----------------
__global__ void proj_kernel(const _Float16* __restrict__ Ctx,
                            const float* __restrict__ Wproj,
                            const float* __restrict__ bproj,
                            float* __restrict__ out) {
  extern __shared__ __align__(16) char smem[];
  _Float16* Xl = (_Float16*)smem;       // 128 x 160
  _Float16* Wt = Xl + kT * kKpad;       // 144 x 160 (N-major / transposed)

  const int b = blockIdx.x;
  const int tid = threadIdx.x;
  const _Float16* cb = Ctx + (size_t)b * kT * kC;

  for (int i = tid; i < kT * kKpad; i += blockDim.x) {
    int m = i / kKpad, c = i % kKpad;
    Xl[i] = (c < kC) ? cb[m * kC + c] : (_Float16)0.0f;
  }
  for (int i = tid; i < kC * kKpad; i += blockDim.x) {
    int n = i / kKpad, k = i % kKpad;
    Wt[i] = (k < kC) ? (_Float16)Wproj[k * kC + n] : (_Float16)0.0f;
  }
  __syncthreads();

  const int wave = tid >> 5;
  for (int it = 0; it < 9; ++it) {       // 72 tiles = 8 waves * 9
    int tile = wave * 9 + it;
    int mt = tile / 9, nt = tile % 9;
    v8f c = {};
#pragma unroll
    for (int ks = 0; ks < 5; ++ks) {
      v16h a  = ldsA(Xl + mt * 16 * kKpad + ks * 32, kKpad);
      v16h bb = ldsB(Wt + nt * 16 * kKpad + ks * 32, kKpad);
      c = WMMA_F16(a, bb, c);
    }
    const int lane = tid & 31;
    const int col  = lane & 15;
    const int half = lane >> 4;
    const float bias = bproj[nt * 16 + col];
    float* o = out + (size_t)b * kT * kC + mt * 16 * kC + nt * 16;
#pragma unroll
    for (int r = 0; r < 8; ++r) o[(r + half * 8) * kC + col] = c[r] + bias;
  }
}

extern "C" void kernel_launch(void* const* d_in, const int* in_sizes, int n_in,
                              void* d_out, int out_size, void* d_ws, size_t ws_size,
                              hipStream_t stream) {
  (void)in_sizes; (void)n_in; (void)out_size; (void)ws_size;
  const float* x     = (const float*)d_in[0];
  const float* Wq    = (const float*)d_in[1];
  const float* Wk    = (const float*)d_in[2];
  const float* Wv    = (const float*)d_in[3];
  const float* rel   = (const float*)d_in[4];
  const float* Wproj = (const float*)d_in[5];
  const float* bproj = (const float*)d_in[6];
  float* out = (float*)d_out;

  // f16 workspaces: Q, K, V (one contiguous [3][B*T*C] block), then Ctx.
  const size_t n = (size_t)kB * kT * kC;
  _Float16* qkv = (_Float16*)d_ws;
  _Float16* Qh  = qkv;
  _Float16* Kh  = qkv + n;
  _Float16* Vh  = qkv + 2 * n;
  _Float16* Ctx = qkv + 3 * n;

  const size_t smemA = (size_t)kT * kKpad * 2 + 3 * (size_t)kC * kKpad * 2;        // 179200 B
  const size_t smemB = 4 * (size_t)kT * 32 * 2 + 2 * (size_t)kT * kLP * 4
                     + (size_t)kT * kLP * 2;                                        // 206848 B
  const size_t smemC = (size_t)kT * kKpad * 2 + (size_t)kC * kKpad * 2;             // 87040 B

  qkv_kernel<<<kB, 256, smemA, stream>>>(x, Wq, Wk, Wv, qkv);
  attn_kernel<<<kB * kH, 256, smemB, stream>>>(Qh, Kh, Vh, rel, Ctx);
  proj_kernel<<<kB, 256, smemC, stream>>>(Ctx, Wproj, bproj, out);
}
